// MusicGeneration_41283225649473
// MI455X (gfx1250) — matrix operand order repaired and checked by reading, not compiled, not run on hardware
//
#include <hip/hip_runtime.h>

typedef __attribute__((ext_vector_type(4)))  float  v4f;
typedef __attribute__((ext_vector_type(8)))  float  v8f;
typedef __attribute__((ext_vector_type(4)))  __bf16 v4bf;
typedef __attribute__((ext_vector_type(8)))  __bf16 v8bf;
typedef __attribute__((ext_vector_type(16))) __bf16 v16bf;

#define NOTES  78
#define IN_DIM 80
#define TH     64          // time-LSTM hidden
#define G4T    256         // 4*TH
#define TSTEPS 128
#define BATCH  64
#define KX     96          // input K padded to multiple of 32
#define KTOT   160         // KX + TH
#define KPAD   168         // LDS row pad (336B rows, 16B aligned)
#define ROWS_T (BATCH*NOTES)   // 4992 = 312*16
#define PDIM   (BATCH*TSTEPS)  // 8192 = 512*16
#define NH     2
#define G4N    8

// branch-free sigmoid/tanh: v_exp_f32 + v_rcp_f32 (both TRANS, co-issue with VALU)
static __device__ __forceinline__ float sigf(float x) {
    return __builtin_amdgcn_rcpf(1.0f + __expf(-x));
}
static __device__ __forceinline__ float tanh_fast(float x) {
    return 1.0f - 2.0f * __builtin_amdgcn_rcpf(__expf(2.0f * x) + 1.0f);
}

// 16 consecutive bf16 (B-fragment chunk): two 16B loads
static __device__ __forceinline__ v16bf frag_contig(const __bf16* p) {
    v8bf lo = *(const v8bf*)(p);
    v8bf hi = *(const v8bf*)(p + 8);
    return __builtin_shufflevector(lo, hi, 0,1,2,3,4,5,6,7,8,9,10,11,12,13,14,15);
}
// A-fragment chunk: [p..p+7] and [p+16..p+23]
static __device__ __forceinline__ v16bf frag_split(const __bf16* p) {
    v8bf lo = *(const v8bf*)(p);
    v8bf hi = *(const v8bf*)(p + 16);
    return __builtin_shufflevector(lo, hi, 0,1,2,3,4,5,6,7,8,9,10,11,12,13,14,15);
}

// Pack [W_ih | 0pad | W_hh] as bf16, [n][k] (k contiguous) for B-fragment loads.
__global__ __launch_bounds__(256) void pack_w_kernel(const float* __restrict__ w_ih,
                                                     const float* __restrict__ w_hh,
                                                     __bf16* __restrict__ wpk) {
    int idx = blockIdx.x * 256 + threadIdx.x;
    if (idx >= G4T * KTOT) return;
    int n = idx / KTOT, k = idx - n * KTOT;
    float v;
    if (k < IN_DIM)      v = w_ih[n * IN_DIM + k];
    else if (k < KX)     v = 0.0f;
    else                 v = w_hh[n * TH + (k - KX)];
    wpk[idx] = (__bf16)v;
}

// Persistent time-LSTM: one workgroup owns 16 (b,note) rows for all 128 steps.
__global__ __launch_bounds__(256) void time_lstm_kernel(
    const float* __restrict__ x,
    const __bf16* __restrict__ wpk,
    const float* __restrict__ b_ih,
    const float* __restrict__ b_hh,
    __bf16* __restrict__ tout)   // [NOTES][PDIM][TH] bf16
{
    __shared__ __align__(16) __bf16 sA[16][KPAD];   // [x(0..79) | 0(80..95) | h(96..159)]
    __shared__ __align__(16) float  sG[16][G4T];    // gate pre-activations

    const int tid   = threadIdx.x;
    const int wave  = tid >> 5;
    const int lane  = tid & 31;
    const int lm    = lane & 15;
    const int khalf = lane >> 4;
    const int R0    = blockIdx.x * 16;

    // ---- x staging: 16x80 floats = 320 float4 chunks; thread -> chunk tid (+256 if tid<64)
    const int  r0c = tid / 20,          d0c = (tid % 20) * 4;
    const bool has2 = (tid < 64);
    const int  c1  = 256 + tid;
    const int  r1c = c1 / 20,           d1c = (c1 % 20) * 4;
    const float* xp0 = x + (size_t)(R0 + r0c) * TSTEPS * IN_DIM + d0c;
    const float* xp1 = x + (size_t)(R0 + r1c) * TSTEPS * IN_DIM + d1c;

    // zero the pad + initial-h region of sA
    for (int idx = tid; idx < 16 * (KPAD - IN_DIM); idx += 256) {
        int r = idx / (KPAD - IN_DIM);
        int k = IN_DIM + (idx - r * (KPAD - IN_DIM));
        sA[r][k] = (__bf16)0.0f;
    }

    // elementwise ownership: thread -> (row, 4 hidden units); c stays in regs
    const int urow = tid >> 4;
    const int j0   = (tid & 15) * 4;
    float creg[4] = {0.f, 0.f, 0.f, 0.f};

    // loop-invariant biases in registers
    float biasi[4], biasf[4], biasg[4], biaso[4];
    #pragma unroll
    for (int q = 0; q < 4; ++q) {
        int j = j0 + q;
        biasi[q] = b_ih[j]          + b_hh[j];
        biasf[q] = b_ih[TH + j]     + b_hh[TH + j];
        biasg[q] = b_ih[2 * TH + j] + b_hh[2 * TH + j];
        biaso[q] = b_ih[3 * TH + j] + b_hh[3 * TH + j];
    }

    const int Rg   = R0 + urow;
    const int bb   = Rg / NOTES;
    const int note = Rg - bb * NOTES;
    __bf16* orow0  = tout + ((size_t)note * PDIM + (size_t)bb * TSTEPS) * TH + j0;

    // prefetch t=0 x tile into registers
    v4f xa = *(const v4f*)(xp0);
    v4f xb = has2 ? *(const v4f*)(xp1) : (v4f)0.0f;

    for (int t = 0; t < TSTEPS; ++t) {
        // commit staged x (bf16) into A layout
        {
            v4bf pa;
            #pragma unroll
            for (int q = 0; q < 4; ++q) pa[q] = (__bf16)xa[q];
            *(v4bf*)(&sA[r0c][d0c]) = pa;
            if (has2) {
                v4bf pb;
                #pragma unroll
                for (int q = 0; q < 4; ++q) pb[q] = (__bf16)xb[q];
                *(v4bf*)(&sA[r1c][d1c]) = pb;
            }
        }
        __syncthreads();

        // software pipeline: issue next step's x loads; latency hides behind WMMA+gates
        if (t + 1 < TSTEPS) {
            xa = *(const v4f*)(xp0 + (size_t)(t + 1) * IN_DIM);
            if (has2) xb = *(const v4f*)(xp1 + (size_t)(t + 1) * IN_DIM);
        }

        // gates = [x|h] @ W : each wave does two 16x16 N-tiles, K=160 = 5 wmma
        #pragma unroll
        for (int i = 0; i < 2; ++i) {
            const int ntile = wave * 2 + i;
            const __bf16* wcol = wpk + (size_t)(ntile * 16 + lm) * KTOT;
            v8f acc = (v8f)0.0f;
            #pragma unroll
            for (int kb = 0; kb < 5; ++kb) {
                v16bf a = frag_split (&sA[lm][0] + kb * 32 + khalf * 8);
                v16bf b = frag_split (wcol       + kb * 32 + khalf * 8);
                acc = __builtin_amdgcn_wmma_f32_16x16x32_bf16(
                          false, a, false, b, (short)0, acc, false, false);
            }
            #pragma unroll
            for (int v = 0; v < 8; ++v)
                sG[v + khalf * 8][ntile * 16 + lm] = acc[v];
        }
        __syncthreads();

        // i,f,g,o -> c,h ; write h back into A staging (bf16) + to workspace
        {
            const v4f gi4 = *(const v4f*)&sG[urow][j0];
            const v4f gf4 = *(const v4f*)&sG[urow][TH + j0];
            const v4f gg4 = *(const v4f*)&sG[urow][2 * TH + j0];
            const v4f go4 = *(const v4f*)&sG[urow][3 * TH + j0];
            v4bf hpk;
            #pragma unroll
            for (int q = 0; q < 4; ++q) {
                float ig = sigf     (gi4[q] + biasi[q]);
                float fg = sigf     (gf4[q] + biasf[q]);
                float gg = tanh_fast(gg4[q] + biasg[q]);
                float og = sigf     (go4[q] + biaso[q]);
                float c  = fg * creg[q] + ig * gg;
                creg[q]  = c;
                hpk[q]   = (__bf16)(og * tanh_fast(c));
            }
            *(v4bf*)(&sA[urow][KX + j0]) = hpk;       // h for next step's A
            *(v4bf*)(orow0 + (size_t)t * TH) = hpk;   // h -> workspace (note-LSTM A layout)
        }
        // next iteration's post-store barrier orders h-write vs wmma reads
    }
}

// Note-LSTM: one wave per 16 (b,t) rows, sequence over 78 notes, H=2.
__global__ __launch_bounds__(32) void note_lstm_kernel(
    const __bf16* __restrict__ tin,   // [NOTES][PDIM][TH] bf16
    const float* __restrict__ w_ih,   // (8,64)
    const float* __restrict__ w_hh,   // (8,2)
    const float* __restrict__ b_ih,
    const float* __restrict__ b_hh,
    float* __restrict__ out)          // [PDIM][156]
{
    __shared__ __align__(16) __bf16 sB[16][TH];     // W_ih_n^T padded to 16 cols
    __shared__ float sG[16][16];
    __shared__ float sH[16][2];

    const int lane  = threadIdx.x;
    const int lm    = lane & 15;
    const int khalf = lane >> 4;
    const int p0    = blockIdx.x * 16;

    for (int idx = lane; idx < 16 * TH; idx += 32) {
        int n = idx / TH, k = idx - n * TH;
        sB[n][k] = (__bf16)((n < G4N) ? w_ih[n * TH + k] : 0.0f);
    }
    ((float*)sH)[lane] = 0.0f;
    __syncthreads();

    // B fragments live in registers for the whole sequence
    v16bf bf0 = frag_split(&sB[lm][0] + khalf * 8);
    v16bf bf1 = frag_split(&sB[lm][0] + 32 + khalf * 8);

    const int row = lm;        // batch row within tile
    const int j   = khalf;     // hidden unit 0/1
    float whh0[4], whh1[4], bia[4];
    #pragma unroll
    for (int q = 0; q < 4; ++q) {
        int k = 2 * q + j;     // gate index for unit j (i,f pairs etc.)
        whh0[q] = w_hh[k * NH + 0];
        whh1[q] = w_hh[k * NH + 1];
        bia[q]  = b_ih[k] + b_hh[k];
    }

    float h = 0.f, c = 0.f;
    for (int note = 0; note < NOTES; ++note) {
        const __bf16* arow = tin + ((size_t)note * PDIM + p0 + lm) * TH;
        v8f acc = (v8f)0.0f;
        v16bf a0 = frag_split(arow + khalf * 8);
        v16bf a1 = frag_split(arow + 32 + khalf * 8);
        acc = __builtin_amdgcn_wmma_f32_16x16x32_bf16(false, a0, false, bf0, (short)0, acc, false, false);
        acc = __builtin_amdgcn_wmma_f32_16x16x32_bf16(false, a1, false, bf1, (short)0, acc, false, false);
        #pragma unroll
        for (int v = 0; v < 8; ++v) sG[v + khalf * 8][lm] = acc[v];
        __syncthreads();

        float h0 = sH[row][0], h1 = sH[row][1];
        float g[4];
        #pragma unroll
        for (int q = 0; q < 4; ++q)
            g[q] = sG[row][2 * q + j] + bia[q] + h0 * whh0[q] + h1 * whh1[q];
        float ig = sigf(g[0]), fg = sigf(g[1]);
        float gg = tanh_fast(g[2]), og = sigf(g[3]);
        c = fg * c + ig * gg;
        h = og * tanh_fast(c);
        out[(size_t)(p0 + row) * (NOTES * NH) + note * NH + j] = (h > 0.5f) ? 1.0f : 0.0f;
        __syncthreads();
        sH[row][j] = h;
        __syncthreads();
    }
}

extern "C" void kernel_launch(void* const* d_in, const int* in_sizes, int n_in,
                              void* d_out, int out_size, void* d_ws, size_t ws_size,
                              hipStream_t stream) {
    (void)in_sizes; (void)n_in; (void)out_size; (void)ws_size;
    const float* x      = (const float*)d_in[0];
    const float* w_ih_t = (const float*)d_in[1];
    const float* w_hh_t = (const float*)d_in[2];
    const float* b_ih_t = (const float*)d_in[3];
    const float* b_hh_t = (const float*)d_in[4];
    const float* w_ih_n = (const float*)d_in[5];
    const float* w_hh_n = (const float*)d_in[6];
    const float* b_ih_n = (const float*)d_in[7];
    const float* b_hh_n = (const float*)d_in[8];

    __bf16* wpk  = (__bf16*)d_ws;                                // 256*160 bf16 = 80 KB
    __bf16* tout = (__bf16*)((char*)d_ws + (size_t)(128 << 10)); // 78*8192*64 bf16 ~ 78 MB

    pack_w_kernel<<<dim3((G4T * KTOT + 255) / 256), dim3(256), 0, stream>>>(w_ih_t, w_hh_t, wpk);
    time_lstm_kernel<<<dim3(ROWS_T / 16), dim3(256), 0, stream>>>(x, wpk, b_ih_t, b_hh_t, tout);
    note_lstm_kernel<<<dim3(PDIM / 16), dim3(32), 0, stream>>>(tout, w_ih_n, w_hh_n, b_ih_n, b_hh_n, (float*)d_out);
}
